// VisionMamba_21715354649055
// MI455X (gfx1250) — matrix operand order repaired and checked by reading, not compiled, not run on hardware
//
#include <hip/hip_runtime.h>

#define BATCH 8
#define E     384
#define DEPTH 24
#define IMG   224
#define PTCH  16
#define LTOK  196          // (224/16)^2
#define DI    768          // 2*E
#define DS    16
#define DCV   4
#define DTR   24
#define NC    1000
#define ROWS  (BATCH * LTOK)   // 1568
#define MPAD  1792             // 7 * 256: GEMM row padding (full blocks, no guards)
#define XPN   64               // x_proj N padded 56 -> 64
#define F2    (2 * DI)         // 1536

typedef __attribute__((ext_vector_type(16))) __bf16 v16bf;
typedef __attribute__((ext_vector_type(8)))  __bf16 v8bf;
typedef __attribute__((ext_vector_type(8)))  float  v8f;

#ifndef __has_builtin
#define __has_builtin(x) 0
#endif
#if __has_builtin(__builtin_amdgcn_global_load_async_to_lds_b128) && \
    __has_builtin(__builtin_amdgcn_s_wait_asynccnt)
#define USE_ASYNC_LDS 1
// Builtin signature (probed): param0 = AS(1) int4*, param1 = AS(3) int4*,
// then two immediate ints (offset, cpol). Vector type is a GENERIC vector.
typedef int gv4i __attribute__((vector_size(16)));
typedef __attribute__((address_space(1))) gv4i as1_v4i;
typedef __attribute__((address_space(3))) gv4i as3_v4i;
#else
#define USE_ASYNC_LDS 0
#endif

__device__ __forceinline__ unsigned short f2bf(float f) {
  unsigned int u = __float_as_uint(f);
  u += 0x7FFFu + ((u >> 16) & 1u);
  return (unsigned short)(u >> 16);
}
__device__ __forceinline__ float siluf(float x) { return x / (1.f + __expf(-x)); }
__device__ __forceinline__ float softplusf(float x) { return x > 20.f ? x : log1pf(__expf(x)); }

// ---------------------------------------------------------------------------
// bf16 WMMA GEMM, LDS-staged + double-buffered async pipeline.
//   C(MPAD x N, f32) = A(MPAD x K, bf16) * W(N x K, bf16)^T
// Block (256 thr, 8 waves) -> 256x64 C tile. Wave w -> 2 m-tiles (32 rows),
// 4 n-tiles; every B-fragment reused by both m-tiles (1.5 ds_loads / WMMA).
// Per 32-K step: A tile 256x32 (16KB) + B tile 64x32 (4KB) staged in LDS.
// Async path: 5x global_load_async_to_lds_b128 per thread, s_wait_asynccnt 5.
// All fragment loads are hoisted ahead of the 8 back-to-back WMMAs.
// ---------------------------------------------------------------------------
__global__ __launch_bounds__(256) void gemm_bf16_wmma(
    const __bf16* __restrict__ A, const __bf16* __restrict__ W,
    float* __restrict__ C, int N, int K)
{
  __shared__ __align__(16) __bf16 sA[2][256 * 32];   // 32 KB
  __shared__ __align__(16) __bf16 sB[2][64 * 32];    //  8 KB

  const int t    = threadIdx.x;
  const int lane = t & 31;
  const int wave = t >> 5;
  const int n64  = N >> 6;
  const int bm   = blockIdx.x / n64;
  const int bn   = blockIdx.x - bm * n64;
  const int m0   = bm * 256;
  const int n0   = bn * 64;

  // Staging: 16B chunks. Chunk i of a tile = (row = i>>2, seg = i&3),
  // LDS linear offset i*8 elements (row-major rows of 32 bf16).
  const int srow = t >> 2;          // 0..63
  const int sseg = (t & 3) * 8;     // 0,8,16,24
  const __bf16* gA0 = A + (size_t)(m0 + srow) * K + sseg;   // rows   0.. 63
  const __bf16* gA1 = gA0 + (size_t)64  * K;                // rows  64..127
  const __bf16* gA2 = gA0 + (size_t)128 * K;                // rows 128..191
  const __bf16* gA3 = gA0 + (size_t)192 * K;                // rows 192..255
  const __bf16* gB  = W + (size_t)(n0 + srow) * K + sseg;

#if USE_ASYNC_LDS
#define STAGE(buf, kofs)                                                        \
  do {                                                                          \
    __builtin_amdgcn_global_load_async_to_lds_b128(                             \
        (as1_v4i*)(gA0 + (kofs)), (as3_v4i*)&sA[buf][t * 8], 0, 0);             \
    __builtin_amdgcn_global_load_async_to_lds_b128(                             \
        (as1_v4i*)(gA1 + (kofs)), (as3_v4i*)&sA[buf][t * 8 + 2048], 0, 0);      \
    __builtin_amdgcn_global_load_async_to_lds_b128(                             \
        (as1_v4i*)(gA2 + (kofs)), (as3_v4i*)&sA[buf][t * 8 + 4096], 0, 0);      \
    __builtin_amdgcn_global_load_async_to_lds_b128(                             \
        (as1_v4i*)(gA3 + (kofs)), (as3_v4i*)&sA[buf][t * 8 + 6144], 0, 0);      \
    __builtin_amdgcn_global_load_async_to_lds_b128(                             \
        (as1_v4i*)(gB + (kofs)), (as3_v4i*)&sB[buf][t * 8], 0, 0);              \
  } while (0)
#else
#define STAGE(buf, kofs)                                                        \
  do {                                                                          \
    *(v8bf*)&sA[buf][t * 8]        = *(const v8bf*)(gA0 + (kofs));              \
    *(v8bf*)&sA[buf][t * 8 + 2048] = *(const v8bf*)(gA1 + (kofs));              \
    *(v8bf*)&sA[buf][t * 8 + 4096] = *(const v8bf*)(gA2 + (kofs));              \
    *(v8bf*)&sA[buf][t * 8 + 6144] = *(const v8bf*)(gA3 + (kofs));              \
    *(v8bf*)&sB[buf][t * 8]        = *(const v8bf*)(gB + (kofs));               \
  } while (0)
#endif

  v8f acc0[4], acc1[4];
  const v8f zero = {0.f,0.f,0.f,0.f,0.f,0.f,0.f,0.f};
#pragma unroll
  for (int i = 0; i < 4; ++i) { acc0[i] = zero; acc1[i] = zero; }

  const int nk = K >> 5;
  STAGE(0, 0);

  for (int ks = 0; ks < nk; ++ks) {
    const int cur = ks & 1;
    if (ks + 1 < nk) {
      STAGE(1 - cur, (ks + 1) * 32);       // next set in flight under WMMAs
#if USE_ASYNC_LDS
      __builtin_amdgcn_s_wait_asynccnt(5); // only wait for set ks (5 ops/thread)
#endif
    } else {
#if USE_ASYNC_LDS
      __builtin_amdgcn_s_wait_asynccnt(0);
#endif
    }
    __syncthreads();

    // A fragments (16-bit 16x32): lane<16 -> row=lane, K={0..15};
    //                             lane>=16 -> same rows, K={16..31}.
    const __bf16* ap = &sA[cur][(wave * 32 + (lane & 15)) * 32 + (lane >> 4) * 16];
    v8bf a0lo = *(const v8bf*)ap;
    v8bf a0hi = *(const v8bf*)(ap + 8);
    v8bf a1lo = *(const v8bf*)(ap + 512);        // m-tile 1: rows +16
    v8bf a1hi = *(const v8bf*)(ap + 520);
    // B fragments: lane&15 = column, (lane>>4)*16 = K half (16 contiguous)
    v16bf bfr[4];
#pragma unroll
    for (int nt = 0; nt < 4; ++nt)
      bfr[nt] = *(const v16bf*)&sB[cur][(nt * 16 + (lane & 15)) * 32 + (lane >> 4) * 16];

    v16bf af0 = __builtin_shufflevector(a0lo, a0hi,
        0,1,2,3,4,5,6,7,8,9,10,11,12,13,14,15);
    v16bf af1 = __builtin_shufflevector(a1lo, a1hi,
        0,1,2,3,4,5,6,7,8,9,10,11,12,13,14,15);

#pragma unroll
    for (int nt = 0; nt < 4; ++nt)
      acc0[nt] = __builtin_amdgcn_wmma_f32_16x16x32_bf16(
          false, af0, false, bfr[nt], (short)0, acc0[nt], false, false);
#pragma unroll
    for (int nt = 0; nt < 4; ++nt)
      acc1[nt] = __builtin_amdgcn_wmma_f32_16x16x32_bf16(
          false, af1, false, bfr[nt], (short)0, acc1[nt], false, false);
    __syncthreads();
  }
#undef STAGE

  // C/D layout: lane&15 = col, VGPR r -> row = r + 8*(lane>>4)
  const int crow = m0 + wave * 32 + 8 * (lane >> 4);
  const int ccol = n0 + (lane & 15);
  float* cp0 = C + (size_t)crow * N + ccol;
  float* cp1 = cp0 + (size_t)16 * N;
#pragma unroll
  for (int r = 0; r < 8; ++r) {
    cp0[(size_t)r * N +  0] = acc0[0][r];
    cp0[(size_t)r * N + 16] = acc0[1][r];
    cp0[(size_t)r * N + 32] = acc0[2][r];
    cp0[(size_t)r * N + 48] = acc0[3][r];
    cp1[(size_t)r * N +  0] = acc1[0][r];
    cp1[(size_t)r * N + 16] = acc1[1][r];
    cp1[(size_t)r * N + 32] = acc1[2][r];
    cp1[(size_t)r * N + 48] = acc1[3][r];
  }
}

// ---------------------------------------------------------------------------
// Residual add + LayerNorm over E=384 (one row per block, 128 threads x 3)
// ---------------------------------------------------------------------------
__global__ __launch_bounds__(128) void add_ln_kernel(
    float* __restrict__ resid, const float* __restrict__ hidden,
    const float* __restrict__ w, const float* __restrict__ b,
    unsigned short* __restrict__ out_bf, float* __restrict__ out_f32)
{
  const int row = blockIdx.x;
  const int tid = threadIdx.x;
  __shared__ float sred[128];
  float v[3];
#pragma unroll
  for (int i = 0; i < 3; ++i) {
    int idx = tid + i * 128;
    float r = resid[(size_t)row * E + idx] + hidden[(size_t)row * E + idx];
    resid[(size_t)row * E + idx] = r;
    v[i] = r;
  }
  sred[tid] = v[0] + v[1] + v[2];
  __syncthreads();
  for (int o = 64; o > 0; o >>= 1) { if (tid < o) sred[tid] += sred[tid + o]; __syncthreads(); }
  const float mu = sred[0] * (1.f / E);
  __syncthreads();
  float d0 = v[0] - mu, d1 = v[1] - mu, d2 = v[2] - mu;
  sred[tid] = d0 * d0 + d1 * d1 + d2 * d2;
  __syncthreads();
  for (int o = 64; o > 0; o >>= 1) { if (tid < o) sred[tid] += sred[tid + o]; __syncthreads(); }
  const float rstd = rsqrtf(sred[0] * (1.f / E) + 1e-5f);
#pragma unroll
  for (int i = 0; i < 3; ++i) {
    int idx = tid + i * 128;
    float y = (v[i] - mu) * rstd * w[idx] + b[idx];
    if (out_bf)  out_bf[(size_t)row * E + idx] = f2bf(y);
    if (out_f32) out_f32[(size_t)row * E + idx] = y;
  }
}

// Causal depthwise conv1d (DCV=4) + SiLU; reads xm = xz[:, :, 0:DI]
__global__ void conv_silu_kernel(
    const float* __restrict__ xz, const float* __restrict__ cw,
    const float* __restrict__ cb, float* __restrict__ xc,
    unsigned short* __restrict__ xc_bf)
{
  int tid = blockIdx.x * blockDim.x + threadIdx.x;
  if (tid >= ROWS * DI) return;
  int d = tid % DI, row = tid / DI;
  int l = row % LTOK, bb = row / LTOK;
  float acc = cb[d];
#pragma unroll
  for (int k = 0; k < DCV; ++k) {
    int ls = l - (DCV - 1) + k;
    if (ls >= 0) acc += cw[d * DCV + k] * xz[(size_t)(bb * LTOK + ls) * F2 + d];
  }
  float y = siluf(acc);
  xc[tid] = y;
  xc_bf[tid] = f2bf(y);
}

// dt = softplus(xdbl[:, :DTR] @ dt_w^T + dt_b), one (b,l) row per block
__global__ __launch_bounds__(DI) void dt_kernel(
    const float* __restrict__ xdbl, const float* __restrict__ dw,
    const float* __restrict__ db, float* __restrict__ dt)
{
  const int row = blockIdx.x;
  const int d = threadIdx.x;
  __shared__ float s[DTR];
  if (d < DTR) s[d] = xdbl[(size_t)row * XPN + d];
  __syncthreads();
  float acc = db[d];
#pragma unroll
  for (int r = 0; r < DTR; ++r) acc += s[r] * dw[d * DTR + r];
  dt[(size_t)row * DI + d] = softplusf(acc);
}

// Selective scan: block = one batch item, thread = channel d; B/C staged in LDS
__global__ __launch_bounds__(DI) void scan_kernel(
    const float* __restrict__ xdbl, const float* __restrict__ dt,
    const float* __restrict__ xc, const float* __restrict__ A_log,
    float* __restrict__ ys)
{
  const int bb = blockIdx.x;
  const int d = threadIdx.x;
  __shared__ float sB[LTOK * DS];
  __shared__ float sC[LTOK * DS];
  for (int i = d; i < LTOK * 2 * DS; i += DI) {
    int l = i >> 5, j = i & 31;
    float v = xdbl[(size_t)(bb * LTOK + l) * XPN + DTR + j];
    if (j < DS) sB[l * DS + j] = v; else sC[l * DS + (j - DS)] = v;
  }
  __syncthreads();
  float Ar[DS], h[DS];
#pragma unroll
  for (int s = 0; s < DS; ++s) { Ar[s] = -__expf(A_log[d * DS + s]); h[s] = 0.f; }
  for (int l = 0; l < LTOK; ++l) {
    size_t idx = (size_t)(bb * LTOK + l) * DI + d;
    float dtv = dt[idx];
    float dtx = dtv * xc[idx];
    float y = 0.f;
#pragma unroll
    for (int s = 0; s < DS; ++s) {
      h[s] = __expf(dtv * Ar[s]) * h[s] + dtx * sB[l * DS + s];
      y += h[s] * sC[l * DS + s];
    }
    ys[idx] = y;
  }
}

// y = (ys + xc*Dp) * silu(z) -> bf16 A-matrix for out_proj
__global__ void gate_kernel(
    const float* __restrict__ ys, const float* __restrict__ xc,
    const float* __restrict__ Dp, const float* __restrict__ xz,
    unsigned short* __restrict__ yg)
{
  int tid = blockIdx.x * blockDim.x + threadIdx.x;
  if (tid >= ROWS * DI) return;
  int d = tid % DI, row = tid / DI;
  float z = xz[(size_t)row * F2 + DI + d];
  yg[tid] = f2bf((ys[tid] + xc[tid] * Dp[d]) * siluf(z));
}

__global__ void pool_kernel(const float* __restrict__ lnf, float* __restrict__ pooled)
{
  int tid = blockIdx.x * blockDim.x + threadIdx.x;
  if (tid >= BATCH * E) return;
  int e = tid % E, bb = tid / E;
  float s = 0.f;
  for (int l = 0; l < LTOK; ++l) s += lnf[(size_t)(bb * LTOK + l) * E + e];
  pooled[tid] = s * (1.f / LTOK);
}

__global__ void head_kernel(
    const float* __restrict__ pooled, const float* __restrict__ hw,
    const float* __restrict__ hb, float* __restrict__ out)
{
  int tid = blockIdx.x * blockDim.x + threadIdx.x;
  if (tid >= BATCH * NC) return;
  int n = tid % NC, bb = tid / NC;
  float acc = hb[n];
  for (int e = 0; e < E; ++e) acc += pooled[bb * E + e] * hw[n * E + e];
  out[tid] = acc;
}

__global__ void cvt_bf16_kernel(const float* __restrict__ src,
                                unsigned short* __restrict__ dst, int n)
{
  int i = blockIdx.x * blockDim.x + threadIdx.x;
  if (i < n) dst[i] = f2bf(src[i]);
}

// x_proj weight: (DEPTH,56,DI) -> bf16 padded to (DEPTH,64,DI), pad rows = 0
__global__ void cvt_xproj_kernel(const float* __restrict__ src,
                                 unsigned short* __restrict__ dst)
{
  int i = blockIdx.x * blockDim.x + threadIdx.x;
  if (i >= DEPTH * XPN * DI) return;
  int k = i % DI, rem = i / DI;
  int n = rem % XPN, layer = rem / XPN;
  dst[i] = (n < DTR + 2 * DS)
             ? f2bf(src[((size_t)layer * (DTR + 2 * DS) + n) * DI + k])
             : (unsigned short)0;
}

// im2col for patch embed: row=(b,l), k=c*256+ky*16+kx, bf16 output
__global__ void im2col_kernel(const float* __restrict__ x,
                              unsigned short* __restrict__ im)
{
  int i = blockIdx.x * blockDim.x + threadIdx.x;
  if (i >= ROWS * 768) return;
  int k = i % 768, row = i / 768;
  int l = row % LTOK, bb = row / LTOK;
  int c = k >> 8, ky = (k >> 4) & 15, kx = k & 15;
  int py = l / 14, px = l % 14;
  im[i] = f2bf(x[((size_t)(bb * 3 + c) * IMG + py * PTCH + ky) * IMG + px * PTCH + kx]);
}

__global__ void bias_zero_kernel(float* __restrict__ hidden,
                                 const float* __restrict__ pb,
                                 float* __restrict__ resid)
{
  int i = blockIdx.x * blockDim.x + threadIdx.x;
  if (i >= ROWS * E) return;
  hidden[i] += pb[i % E];
  resid[i] = 0.f;
}

// ---------------------------------------------------------------------------
extern "C" void kernel_launch(void* const* d_in, const int* in_sizes, int n_in,
                              void* d_out, int out_size, void* d_ws, size_t ws_size,
                              hipStream_t stream)
{
  (void)in_sizes; (void)n_in; (void)out_size; (void)ws_size;
  const float* x       = (const float*)d_in[0];
  const float* patch_w = (const float*)d_in[1];
  const float* patch_b = (const float*)d_in[2];
  const float* norm_w  = (const float*)d_in[3];
  const float* norm_b  = (const float*)d_in[4];
  const float* in_w    = (const float*)d_in[5];
  const float* conv_w  = (const float*)d_in[6];
  const float* conv_b  = (const float*)d_in[7];
  const float* xproj_w = (const float*)d_in[8];
  const float* dt_w    = (const float*)d_in[9];
  const float* dt_b    = (const float*)d_in[10];
  const float* A_log   = (const float*)d_in[11];
  const float* Dp      = (const float*)d_in[12];
  const float* out_w   = (const float*)d_in[13];
  const float* normf_w = (const float*)d_in[14];
  const float* normf_b = (const float*)d_in[15];
  const float* head_w  = (const float*)d_in[16];
  const float* head_b  = (const float*)d_in[17];
  float* out = (float*)d_out;

  char* wsp = (char*)d_ws;
  size_t off = 0;
  auto alloc = [&](size_t bytes) -> char* {
    char* p = wsp + off;
    off = (off + bytes + 255) & ~(size_t)255;
    return p;
  };

  // weights (bf16)
  unsigned short* w_in_bf  = (unsigned short*)alloc((size_t)DEPTH * F2 * E * 2);
  unsigned short* w_out_bf = (unsigned short*)alloc((size_t)DEPTH * E * DI * 2);
  unsigned short* w_xp_bf  = (unsigned short*)alloc((size_t)DEPTH * XPN * DI * 2);
  unsigned short* w_pw_bf  = (unsigned short*)alloc((size_t)E * 768 * 2);
  // GEMM A/C buffers use MPAD rows (padded rows: garbage-in/garbage-out)
  unsigned short* im2col   = (unsigned short*)alloc((size_t)MPAD * 768 * 2);
  float*          resid    = (float*)alloc((size_t)ROWS * E * 4);
  float*          hidden   = (float*)alloc((size_t)MPAD * E * 4);
  unsigned short* hn_bf    = (unsigned short*)alloc((size_t)MPAD * E * 2);
  float*          xz       = (float*)alloc((size_t)MPAD * F2 * 4);
  float*          xc       = (float*)alloc((size_t)ROWS * DI * 4);
  unsigned short* xc_bf    = (unsigned short*)alloc((size_t)MPAD * DI * 2);
  float*          xdbl     = (float*)alloc((size_t)MPAD * XPN * 4);
  float*          dtb      = (float*)alloc((size_t)ROWS * DI * 4);
  float*          ysb      = (float*)alloc((size_t)ROWS * DI * 4);
  unsigned short* yg_bf    = (unsigned short*)alloc((size_t)MPAD * DI * 2);
  float*          lnf      = (float*)alloc((size_t)ROWS * E * 4);
  float*          pooled   = (float*)alloc((size_t)BATCH * E * 4);

  const int T = 256;
  auto cdiv = [](int a, int b) { return (a + b - 1) / b; };
  auto gemm_grid = [](int N) { return (MPAD / 256) * (N >> 6); };

  // ---- weight conversion (re-done every call: deterministic, cheap) ----
  { int n = DEPTH * F2 * E;
    cvt_bf16_kernel<<<cdiv(n, T), T, 0, stream>>>(in_w, w_in_bf, n); }
  { int n = DEPTH * E * DI;
    cvt_bf16_kernel<<<cdiv(n, T), T, 0, stream>>>(out_w, w_out_bf, n); }
  { int n = E * 768;
    cvt_bf16_kernel<<<cdiv(n, T), T, 0, stream>>>(patch_w, w_pw_bf, n); }
  cvt_xproj_kernel<<<cdiv(DEPTH * XPN * DI, T), T, 0, stream>>>(xproj_w, w_xp_bf);

  // ---- patch embed: im2col + WMMA GEMM + bias (resid zeroed alongside) ----
  im2col_kernel<<<cdiv(ROWS * 768, T), T, 0, stream>>>(x, im2col);
  gemm_bf16_wmma<<<gemm_grid(E), T, 0, stream>>>(
      (const __bf16*)im2col, (const __bf16*)w_pw_bf, hidden, E, 768);
  bias_zero_kernel<<<cdiv(ROWS * E, T), T, 0, stream>>>(hidden, patch_b, resid);

  // ---- 24 Mamba layers ----
  for (int layer = 0; layer < DEPTH; ++layer) {
    const __bf16* wi = (const __bf16*)(w_in_bf  + (size_t)layer * F2 * E);
    const __bf16* wx = (const __bf16*)(w_xp_bf  + (size_t)layer * XPN * DI);
    const __bf16* wo = (const __bf16*)(w_out_bf + (size_t)layer * E * DI);

    add_ln_kernel<<<ROWS, 128, 0, stream>>>(
        resid, hidden, norm_w + (size_t)layer * E, norm_b + (size_t)layer * E,
        hn_bf, (float*)nullptr);

    gemm_bf16_wmma<<<gemm_grid(F2), T, 0, stream>>>(
        (const __bf16*)hn_bf, wi, xz, F2, E);

    conv_silu_kernel<<<cdiv(ROWS * DI, T), T, 0, stream>>>(
        xz, conv_w + (size_t)layer * DI * DCV, conv_b + (size_t)layer * DI,
        xc, xc_bf);

    gemm_bf16_wmma<<<gemm_grid(XPN), T, 0, stream>>>(
        (const __bf16*)xc_bf, wx, xdbl, XPN, DI);

    dt_kernel<<<ROWS, DI, 0, stream>>>(
        xdbl, dt_w + (size_t)layer * DI * DTR, dt_b + (size_t)layer * DI, dtb);

    scan_kernel<<<BATCH, DI, 0, stream>>>(
        xdbl, dtb, xc, A_log + (size_t)layer * DI * DS, ysb);

    gate_kernel<<<cdiv(ROWS * DI, T), T, 0, stream>>>(
        ysb, xc, Dp + (size_t)layer * DI, xz, yg_bf);

    gemm_bf16_wmma<<<gemm_grid(E), T, 0, stream>>>(
        (const __bf16*)yg_bf, wo, hidden, E, DI);
  }

  // ---- final residual + LN (fp32 out), pool, head ----
  add_ln_kernel<<<ROWS, 128, 0, stream>>>(
      resid, hidden, normf_w, normf_b, (unsigned short*)nullptr, lnf);
  pool_kernel<<<cdiv(BATCH * E, T), T, 0, stream>>>(lnf, pooled);
  head_kernel<<<cdiv(BATCH * NC, T), T, 0, stream>>>(pooled, head_w, head_b, out);
}
